// InstantNGP_77326591197416
// MI455X (gfx1250) — matrix operand order repaired and checked by reading, not compiled, not run on hardware
//
#include <hip/hip_runtime.h>
#include <hip/hip_bf16.h>

typedef __attribute__((ext_vector_type(16))) _Float16 v16h;
typedef __attribute__((ext_vector_type(8)))  float    v8f;

#define N_SAMPLES (1 << 21)
#define N_RAYS    (128 * 128)
#define TABLE_LOG 19
#define TABLE_SIZE (1u << TABLE_LOG)
#define NUM_LEVELS 12
#define P1 2654435761u
#define P2 805459861u
#define FLTMAXV 3.402823466e38f

union AFrag { v16h v; _Float16 h[16]; };
union CFrag { v8f  v; float    f[8];  };

__device__ __forceinline__ unsigned pk2h(float a, float b) {
  union { _Float16 h[2]; unsigned u; } p;
  p.h[0] = (_Float16)a; p.h[1] = (_Float16)b;   // lowers to v_cvt_pk_f16_f32
  return p.u;
}

// ------- fused hash-grid encode: shared indices, float2 gathers, packed b32 stores -------
__device__ __forceinline__ void hash_encode2(const float x01[3],
                                             const float* __restrict__ tabD,
                                             const float* __restrict__ tabC,
                                             unsigned* __restrict__ fD32,
                                             unsigned* __restrict__ fC32) {
  const unsigned mask = TABLE_SIZE - 1u;
  for (int l = 0; l < NUM_LEVELS; ++l) {
    const float resm1 = (float)((16 << l) - 1);
    unsigned p0[3]; float fr[3];
#pragma unroll
    for (int c = 0; c < 3; ++c) {
      float p  = x01[c] * resm1;
      float fl = floorf(p);
      p0[c] = (unsigned)(int)fl;            // x01 >= 0 so fl >= 0
      fr[c] = p - fl;
    }
    // hash terms for corner 0/1 along each axis (2 multiplies per level total)
    const unsigned hx0 = p0[0],      hx1 = hx0 + 1u;
    const unsigned hy0 = p0[1] * P1, hy1 = hy0 + P1;
    const unsigned hz0 = p0[2] * P2, hz1 = hz0 + P2;
    const float wx1 = fr[0], wx0 = 1.f - wx1;
    const float wy1 = fr[1], wy0 = 1.f - wy1;
    const float wz1 = fr[2], wz0 = 1.f - wz1;
    float d0 = 0.f, d1 = 0.f, c0 = 0.f, c1 = 0.f;
#pragma unroll
    for (int corner = 0; corner < 8; ++corner) {   // 16 independent b64 gathers in flight
      unsigned idx = (((corner & 1) ? hx1 : hx0)
                    ^ ((corner & 2) ? hy1 : hy0)
                    ^ ((corner & 4) ? hz1 : hz0)) & mask;
      float w = ((corner & 1) ? wx1 : wx0)
              * ((corner & 2) ? wy1 : wy0)
              * ((corner & 4) ? wz1 : wz0);
      size_t off = ((size_t)l << TABLE_LOG) + idx;   // L2-resident tables
      float2 ed = ((const float2*)tabD)[off];
      float2 ec = ((const float2*)tabC)[off];
      d0 = fmaf(w, ed.x, d0);  d1 = fmaf(w, ed.y, d1);
      c0 = fmaf(w, ec.x, c0);  c1 = fmaf(w, ec.y, c1);
    }
    fD32[l] = pk2h(d0, d1);
    fC32[l] = pk2h(c0, c1);
  }
#pragma unroll
  for (int l = NUM_LEVELS; l < 16; ++l) {       // pad K 24 -> 32
    fD32[l] = 0u;
    fC32[l] = 0u;
  }
}

// -------- fragment builders, contiguous k = 16*hi + e (wave32, 16-bit A 16x32 / B 32x16) --------
__device__ __forceinline__ v16h load_a_frag(const _Float16* __restrict__ featBase, int lane) {
  AFrag u;
  const int M  = lane & 15;
  const int hi = (lane >> 4) & 1;
#pragma unroll
  for (int e = 0; e < 16; ++e) {
    int j = e >> 1, b = e & 1;
    int K = (hi ? 16 : 0) + 2 * j + b;
    u.h[e] = featBase[M * 32 + K];
  }
  return u.v;
}

__device__ __forceinline__ v16h load_b_w1(const _Float16* __restrict__ w1, int lane, int ncol0) {
  AFrag u;
  const int N  = (lane & 15) + ncol0;     // w1 staged in LDS as f16 [24][32] row-major
  const int hi = (lane >> 4) & 1;
#pragma unroll
  for (int e = 0; e < 16; ++e) {
    int j = e >> 1, b = e & 1;
    int K = (hi ? 16 : 0) + 2 * j + b;
    u.h[e] = (K < 24) ? w1[K * 32 + N] : (_Float16)0.f;
  }
  return u.v;
}

__device__ __forceinline__ v16h load_b_w2(const _Float16* __restrict__ w2, int lane, int outd) {
  AFrag u;
  const int N  = lane & 15;               // w2 staged in LDS as f16 [32][outd], pad cols
  const int hi = (lane >> 4) & 1;
#pragma unroll
  for (int e = 0; e < 16; ++e) {
    int j = e >> 1, b = e & 1;
    int K = (hi ? 16 : 0) + 2 * j + b;
    u.h[e] = (N < outd) ? w2[K * outd + N] : (_Float16)0.f;
  }
  return u.v;
}

__device__ __forceinline__ void store_hidden_relu(_Float16* __restrict__ hidBase, int lane,
                                                  v8f c0, v8f c1) {
  CFrag a, b; a.v = c0; b.v = c1;
  const int N  = lane & 15;
  const int Mb = ((lane >> 4) & 1) ? 8 : 0;   // C/D f32 16x16 layout
#pragma unroll
  for (int r = 0; r < 8; ++r) {
    int M = Mb + r;
    // relu as single v_med3_num_f32 (avoids canonicalize+max pair)
    hidBase[M * 32 + N]      = (_Float16)__builtin_amdgcn_fmed3f(a.f[r], 0.f, FLTMAXV);
    hidBase[M * 32 + N + 16] = (_Float16)__builtin_amdgcn_fmed3f(b.f[r], 0.f, FLTMAXV);
  }
}

// ---------------- kernel 1: encode + MLP for every sample ----------------
__global__ void __launch_bounds__(256) ngp_sample_kernel(
    const float* __restrict__ rays_o, const float* __restrict__ rays_d,
    const float* __restrict__ t_starts, const float* __restrict__ t_ends,
    const int* __restrict__ ray_indices,
    const float* __restrict__ table_d, const float* __restrict__ table_c,
    const float* __restrict__ w_d1, const float* __restrict__ w_d2,
    const float* __restrict__ w_c1, const float* __restrict__ w_c2,
    float* __restrict__ s_out, float* __restrict__ rgb_out) {
  __shared__ unsigned featD[256 * 16];   // 16x uint = 32x f16 per sample row
  __shared__ unsigned featC[256 * 16];
  __shared__ _Float16 hidD[256 * 32];
  __shared__ _Float16 hidC[256 * 32];
  __shared__ _Float16 lw1[2][24 * 32];   // [0]=density, [1]=color
  __shared__ _Float16 lw2d[32];
  __shared__ _Float16 lw2c[32 * 3];
  __shared__ float    dts[256];

  const int tid = threadIdx.x;
  const int i   = blockIdx.x * 256 + tid;

  // warm the weight lines, then stage them to LDS as f16 (once per block)
  __builtin_prefetch(w_d1 + tid, 0, 1);
  __builtin_prefetch(w_c1 + tid, 0, 1);
  for (int k = tid; k < 24 * 32; k += 256) {
    lw1[0][k] = (_Float16)w_d1[k];
    lw1[1][k] = (_Float16)w_c1[k];
  }
  if (tid < 32) lw2d[tid] = (_Float16)w_d2[tid];
  if (tid < 96) lw2c[tid] = (_Float16)w_c2[tid];

  // ---- phase A: per-sample position + fused hash encode (gather-bound, L2-resident) ----
  {
    float ts = t_starts[i], te = t_ends[i];
    dts[tid] = te - ts;
    float tm = 0.5f * (ts + te);
    int   r  = ray_indices[i];
    float x01[3];
#pragma unroll
    for (int c = 0; c < 3; ++c) {
      float x = fmaf(rays_d[r * 3 + c], tm, rays_o[r * 3 + c]);
      x01[c]  = fminf(fmaxf((x + 1.f) * 0.5f, 0.f), 1.f);
    }
    hash_encode2(x01, table_d, table_c, &featD[tid * 16], &featC[tid * 16]);
  }
  __syncthreads();

  // ---- phase B: per-wave WMMA MLP, both branches interleaved to hide WMMA hazards ----
  const int lane = tid & 31;
  const int wave = tid >> 5;
  v16h bD1a = load_b_w1(lw1[0], lane, 0);
  v16h bD1b = load_b_w1(lw1[0], lane, 16);
  v16h bD2  = load_b_w2(lw2d,  lane, 1);
  v16h bC1a = load_b_w1(lw1[1], lane, 0);
  v16h bC1b = load_b_w1(lw1[1], lane, 16);
  v16h bC2  = load_b_w2(lw2c,  lane, 3);
  const v8f z = {};

#pragma unroll
  for (int t = 0; t < 2; ++t) {
    const int rowBase = wave * 32 + t * 16;

    // layer 1, both branches: 4 independent WMMAs back-to-back
    v16h Ad = load_a_frag((const _Float16*)featD + rowBase * 32, lane);
    v16h Ac = load_a_frag((const _Float16*)featC + rowBase * 32, lane);
    v8f d0 = __builtin_amdgcn_wmma_f32_16x16x32_f16(false, Ad, false, bD1a, (short)0, z, false, false);
    v8f d1 = __builtin_amdgcn_wmma_f32_16x16x32_f16(false, Ad, false, bD1b, (short)0, z, false, false);
    v8f e0 = __builtin_amdgcn_wmma_f32_16x16x32_f16(false, Ac, false, bC1a, (short)0, z, false, false);
    v8f e1 = __builtin_amdgcn_wmma_f32_16x16x32_f16(false, Ac, false, bC1b, (short)0, z, false, false);

    // relu + C/D->A transpose through LDS, both branches
    store_hidden_relu(&hidD[rowBase * 32], lane, d0, d1);
    store_hidden_relu(&hidC[rowBase * 32], lane, e0, e1);
    asm volatile("s_wait_dscnt 0" ::: "memory");   // cross-lane LDS RAW

    // layer 2, both branches
    v16h A2d = load_a_frag(&hidD[rowBase * 32], lane);
    v16h A2c = load_a_frag(&hidC[rowBase * 32], lane);
    v8f sd = __builtin_amdgcn_wmma_f32_16x16x32_f16(false, A2d, false, bD2, (short)0, z, false, false);
    v8f cc = __builtin_amdgcn_wmma_f32_16x16x32_f16(false, A2c, false, bC2, (short)0, z, false, false);

    // sigma = exp(head) * dt   (column 0 of the density D tile)
    if ((lane & 15) == 0) {
      CFrag u; u.v = sd;
      const int Mb = (lane >> 4) ? 8 : 0;
#pragma unroll
      for (int r = 0; r < 8; ++r) {
        int row = rowBase + Mb + r;
        s_out[blockIdx.x * 256 + row] = __expf(u.f[r]) * dts[row];
      }
    }
    // rgb = sigmoid(head[:, 0:3])
    const int n = lane & 15;
    if (n < 3) {
      CFrag u; u.v = cc;
      const int Mb = (lane >> 4) ? 8 : 0;
#pragma unroll
      for (int r = 0; r < 8; ++r) {
        int si = blockIdx.x * 256 + rowBase + Mb + r;
        rgb_out[si * 3 + n] = 1.f / (1.f + __expf(-u.f[r]));
      }
    }
  }
}

// ---------------- kernel 2: wave-per-ray transmittance scan ----------------
__device__ __forceinline__ int lower_bound_i(const int* __restrict__ a, int n, int key) {
  int lo = 0, hi = n;
  while (lo < hi) { int mid = (lo + hi) >> 1; if (a[mid] < key) lo = mid + 1; else hi = mid; }
  return lo;
}

__global__ void __launch_bounds__(256) ngp_render_kernel(
    const int* __restrict__ ray_indices, const float* __restrict__ s,
    const float* __restrict__ rgb, float* __restrict__ out, int nsamples) {
  const int lane = threadIdx.x & 31;
  const int ray  = blockIdx.x * 8 + (threadIdx.x >> 5);

  // ray_indices is sorted: segment = [lo, hi)
  const int lo = lower_bound_i(ray_indices, nsamples, ray);
  const int hi = lower_bound_i(ray_indices, nsamples, ray + 1);

  float T = 1.f, cr = 0.f, cg = 0.f, cb = 0.f, acc = 0.f;
  for (int base = lo; base < hi; base += 32) {
    int  idx   = base + lane;
    bool valid = idx < hi;
    float sv = valid ? s[idx] : 0.f;
    float sc = sv;                           // wave-inclusive scan of s
#pragma unroll
    for (int off = 1; off < 32; off <<= 1) {
      float o = __shfl_up(sc, off, 32);
      if (lane >= off) sc += o;
    }
    float w = valid ? (1.f - __expf(-sv)) * T * __expf(-(sc - sv)) : 0.f;
    if (valid) {
      cr  += w * rgb[idx * 3 + 0];
      cg  += w * rgb[idx * 3 + 1];
      cb  += w * rgb[idx * 3 + 2];
      acc += w;
    }
    T *= __expf(-__shfl(sc, 31, 32));        // carry transmittance to next chunk
  }
#pragma unroll
  for (int off = 16; off > 0; off >>= 1) {
    cr  += __shfl_down(cr, off, 32);
    cg  += __shfl_down(cg, off, 32);
    cb  += __shfl_down(cb, off, 32);
    acc += __shfl_down(acc, off, 32);
  }
  if (lane == 0) {                           // CHW color, then alpha
    float bgw = 1.f - acc;                   // BG_COLOR = 1.0
    out[0 * N_RAYS + ray] = fminf(fmaxf(cr + bgw, 0.f), 1.f);
    out[1 * N_RAYS + ray] = fminf(fmaxf(cg + bgw, 0.f), 1.f);
    out[2 * N_RAYS + ray] = fminf(fmaxf(cb + bgw, 0.f), 1.f);
    out[3 * N_RAYS + ray] = fminf(fmaxf(acc, 0.f), 1.f);
  }
}

extern "C" void kernel_launch(void* const* d_in, const int* in_sizes, int n_in,
                              void* d_out, int out_size, void* d_ws, size_t ws_size,
                              hipStream_t stream) {
  const float* rays_o      = (const float*)d_in[0];
  const float* rays_d      = (const float*)d_in[1];
  const float* t_starts    = (const float*)d_in[2];
  const float* t_ends      = (const float*)d_in[3];
  const int*   ray_indices = (const int*)d_in[4];
  const float* table_d     = (const float*)d_in[5];
  const float* table_c     = (const float*)d_in[6];
  const float* w_d1        = (const float*)d_in[7];
  const float* w_d2        = (const float*)d_in[8];
  const float* w_c1        = (const float*)d_in[9];
  const float* w_c2        = (const float*)d_in[10];

  float* s   = (float*)d_ws;           // [N_SAMPLES]
  float* rgb = s + N_SAMPLES;          // [N_SAMPLES * 3]

  ngp_sample_kernel<<<N_SAMPLES / 256, 256, 0, stream>>>(
      rays_o, rays_d, t_starts, t_ends, ray_indices,
      table_d, table_c, w_d1, w_d2, w_c1, w_c2, s, rgb);

  ngp_render_kernel<<<N_RAYS / 8, 256, 0, stream>>>(
      ray_indices, s, rgb, (float*)d_out, N_SAMPLES);
}
